// metaCLF_20504173871859
// MI455X (gfx1250) — compile-verified, matrix operand order checked by reading, not run on hardware
//
#include <hip/hip_runtime.h>
#include <hip/hip_bf16.h>

typedef _Float16 half_t;
typedef __attribute__((ext_vector_type(2)))  _Float16 h2;
typedef __attribute__((ext_vector_type(16))) _Float16 v16h;
typedef __attribute__((ext_vector_type(8)))  float    v8f;

#define NVOX (96*96*96)           // 884736 voxels per batch image
#define NCH  32
#define NTILES ((2*NVOX)/32)      // 55296 wave-tiles of 32 voxels

__device__ __forceinline__ v8f wmma16(v16h a, v16h b, v8f c) {
  // D(f32 16x16) = A(f16 16x32) * B(f16 32x16) + C
  return __builtin_amdgcn_wmma_f32_16x16x32_f16(false, a, false, b, (short)0, c,
                                                false, false);
}

// Build one 16x32 f16 A-fragment of row-major weight matrix W (stride floats/row).
// Rows: rowBase + (lane&15).  K: kBase + kk + (lane>=16 ? 8 : 0) per ISA layout.
__device__ __forceinline__ v16h load_afrag(const float* __restrict__ W, int stride,
                                           int kValid, int rowBase, int kBase, int lane) {
  const int m  = rowBase + (lane & 15);
  const int kh = (lane & 16) ? 8 : 0;
  v16h a;
#pragma unroll
  for (int j = 0; j < 16; ++j) {
    const int kk = ((j & 8) << 1) | (j & 7);          // 0..7, 16..23
    const int k  = kBase + kk + kh;
    const float v = (k < kValid) ? W[m * stride + k] : 0.0f;
    a[j] = (_Float16)v;
  }
  return a;
}

// f32->f16 pack (fuses into v_cvt_pk_f16_f32) then packed-f16 relu (v_pk_max).
// D-fragments of M-tiles (t, t+1) map element-local onto the B-fragment K-layout
// for rows 16t..16t+31 of the next GEMM (no cross-lane movement needed).
__device__ __forceinline__ v16h pack_relu(v8f lo, v8f hi) {
  v16h r;
#pragma unroll
  for (int j = 0; j < 8; ++j) r[j]     = (_Float16)lo[j];
#pragma unroll
  for (int j = 0; j < 8; ++j) r[8 + j] = (_Float16)hi[j];
  const v16h z = {};
  return __builtin_elementwise_max(r, z);
}

__global__ __launch_bounds__(256)
void metaclf_wmma_kernel(const float* __restrict__ x, const float* __restrict__ d_all,
                         const float* __restrict__ W1, const float* __restrict__ W2,
                         const float* __restrict__ G1, const float* __restrict__ G2,
                         const float* __restrict__ G3, float* __restrict__ out) {
  __shared__ float  s_w12[80];             // W1 (32) | W2 (48)
  __shared__ half_t s_enc[8][32 * 36];     // per-wave [voxel n][k<32] pane, 72B pitch

  const int tid = threadIdx.x;
  if (tid < 32)      s_w12[tid] = W1[tid];
  else if (tid < 80) s_w12[tid] = W2[tid - 32];
  __syncthreads();

  const int lane = tid & 31;
  const int wave = tid >> 5;
  const int kh   = (lane & 16) ? 8 : 0;    // K half-select / D-fragment row offset
  half_t* enc = &s_enc[wave][0];

  // Stationary weight A-fragments (10 x v16h = 80 VGPRs, loaded once per wave).
  v16h aG1[2], aG2[4], aG30[2], aG31[2];
#pragma unroll
  for (int t = 0; t < 2; ++t) aG1[t] = load_afrag(G1, 27, 27, 16 * t, 0, lane);
#pragma unroll
  for (int t = 0; t < 4; ++t) aG2[t] = load_afrag(G2, 32, 32, 16 * t, 0, lane);
#pragma unroll
  for (int t = 0; t < 2; ++t) {
    aG30[t] = load_afrag(G3, 64, 64, 16 * t, 0,  lane);
    aG31[t] = load_afrag(G3, 64, 64, 16 * t, 32, lane);
  }

  // Packed-f16 encode weights in registers (40 dwords): pair i2 covers hidden
  // units (2*i2, 2*i2+1).  W1[i][j] = s_w12[2i+j];  W2[c][i] = s_w12[32+16c+i].
  h2 w1a[8], w1b[8], w2p[3][8];
#pragma unroll
  for (int i2 = 0; i2 < 8; ++i2) {
    w1a[i2] = (h2){(_Float16)s_w12[4*i2 + 0], (_Float16)s_w12[4*i2 + 2]};
    w1b[i2] = (h2){(_Float16)s_w12[4*i2 + 1], (_Float16)s_w12[4*i2 + 3]};
#pragma unroll
    for (int c = 0; c < 3; ++c)
      w2p[c][i2] = (h2){(_Float16)s_w12[32 + c*16 + 2*i2],
                        (_Float16)s_w12[32 + c*16 + 2*i2 + 1]};
  }

  const v8f vzero = {};
  const h2  hzero = {};

  const int waveGlobal = (blockIdx.x * blockDim.x + tid) >> 5;
  const int wavesTotal = (gridDim.x * blockDim.x) >> 5;

  for (int t = waveGlobal; t < NTILES; t += wavesTotal) {
    const long vg = (long)t * 32;                  // tiles never straddle a batch (NVOX%32==0)
    const int  b  = (int)(vg / NVOX);
    const long v0 = vg - (long)b * (long)NVOX;
    const float* db = d_all + (long)b * 6   * NVOX + v0;
    const float* xb = x     + (long)b * NCH * NVOX + v0;

    // ---- issue the dominant global traffic (x) early: overlaps the WMMA chain ----
    float xv[2][16];
#pragma unroll
    for (int s = 0; s < 2; ++s) {
      const float* xn = xb + (lane & 15) + 16 * s;
#pragma unroll
      for (int r = 0; r < 8; ++r) {
        xv[s][r]     = xn[(long)(r + kh)      * NVOX];
        xv[s][8 + r] = xn[(long)(16 + r + kh) * NVOX];
      }
    }

    // ---- encode in packed f16: lane == local voxel; coalesced d_all loads ----
    h2 dp[6];
#pragma unroll
    for (int c = 0; c < 6; ++c) {
      const _Float16 h = (_Float16)db[(long)c * NVOX + lane];
      dp[c] = (h2){h, h};
    }

    _Float16 ax[3][3];                              // ax[axis][c]
#pragma unroll
    for (int a = 0; a < 3; ++a) {
      h2 acc0 = hzero, acc1 = hzero, acc2 = hzero;
#pragma unroll
      for (int i2 = 0; i2 < 8; ++i2) {
        h2 e = w1a[i2] * dp[2*a] + w1b[i2] * dp[2*a + 1];   // v_pk_mul + v_pk_fma
        e = __builtin_elementwise_max(e, hzero);            // v_pk_max (relu)
        acc0 += w2p[0][i2] * e;                             // v_pk_fma x3
        acc1 += w2p[1][i2] * e;
        acc2 += w2p[2][i2] * e;
      }
      ax[a][0] = acc0[0] + acc0[1];
      ax[a][1] = acc1[0] + acc1[1];
      ax[a][2] = acc2[0] + acc2[1];
    }

    // enced = kron(kron(x_x,y_y),z_z): idx = i*9 + j*3 + k ; pad K to 32 with 0
    _Float16 eyz[9];
#pragma unroll
    for (int j = 0; j < 3; ++j)
#pragma unroll
      for (int k = 0; k < 3; ++k)
        eyz[j*3 + k] = ax[1][j] * ax[2][k];

    half_t* row = enc + lane * 36;
#pragma unroll
    for (int i = 0; i < 3; ++i)
#pragma unroll
      for (int jk = 0; jk < 9; ++jk)
        row[i*9 + jk] = ax[0][i] * eyz[jk];
#pragma unroll
    for (int k = 27; k < 32; ++k) row[k] = (_Float16)0.0f;

    asm volatile("s_wait_dscnt 0x0" ::: "memory");  // wave-local LDS RAW fence

    // ---- dual-subtile pipeline over the three WMMA GEMM stages ----
    v16h bE[2];                                     // B-fragments of enced [K=32, N=16]
#pragma unroll
    for (int s = 0; s < 2; ++s) {
      const int n = (lane & 15) + 16 * s;
      const half_t* rn = enc + n * 36 + kh;
#pragma unroll
      for (int v = 0; v < 8; ++v) {
        const int k0 = (v < 4) ? (2 * v) : (8 + 2 * v);   // 0,2,4,6,16,18,20,22
        bE[s][2*v]   = rn[k0];
        bE[s][2*v+1] = rn[k0 + 1];
      }
    }

    // h1 = relu(G1 * enced)  [32x16]
    v8f h1lo[2], h1hi[2];
#pragma unroll
    for (int s = 0; s < 2; ++s) h1lo[s] = wmma16(aG1[0], bE[s], vzero);
#pragma unroll
    for (int s = 0; s < 2; ++s) h1hi[s] = wmma16(aG1[1], bE[s], vzero);
    v16h bH1[2];
#pragma unroll
    for (int s = 0; s < 2; ++s) bH1[s] = pack_relu(h1lo[s], h1hi[s]);

    // h2 = relu(G2 * h1)     [64x16]
    v8f e0[2], e1[2], e2[2], e3[2];
#pragma unroll
    for (int s = 0; s < 2; ++s) e0[s] = wmma16(aG2[0], bH1[s], vzero);
#pragma unroll
    for (int s = 0; s < 2; ++s) e1[s] = wmma16(aG2[1], bH1[s], vzero);
#pragma unroll
    for (int s = 0; s < 2; ++s) e2[s] = wmma16(aG2[2], bH1[s], vzero);
#pragma unroll
    for (int s = 0; s < 2; ++s) e3[s] = wmma16(aG2[3], bH1[s], vzero);
    v16h bLo[2], bHi[2];
#pragma unroll
    for (int s = 0; s < 2; ++s) bLo[s] = pack_relu(e0[s], e1[s]);   // K 0..31 of h2
#pragma unroll
    for (int s = 0; s < 2; ++s) bHi[s] = pack_relu(e2[s], e3[s]);   // K 32..63 of h2

    // filters = G3 * h2      [32x16], f32 accumulation across the two K-tiles
    v8f f0[2], f1[2];
#pragma unroll
    for (int s = 0; s < 2; ++s) f0[s] = wmma16(aG30[0], bLo[s], vzero);
#pragma unroll
    for (int s = 0; s < 2; ++s) f1[s] = wmma16(aG30[1], bLo[s], vzero);
#pragma unroll
    for (int s = 0; s < 2; ++s) f0[s] = wmma16(aG31[0], bHi[s], f0[s]);
#pragma unroll
    for (int s = 0; s < 2; ++s) f1[s] = wmma16(aG31[1], bHi[s], f1[s]);

    // out[n] = sum_c filters[c][n] * x[c][n]; lane holds channels kh+r / 16+kh+r
#pragma unroll
    for (int s = 0; s < 2; ++s) {
      float acc = 0.0f;
#pragma unroll
      for (int r = 0; r < 8; ++r) {
        acc += f0[s][r] * xv[s][r];
        acc += f1[s][r] * xv[s][8 + r];
      }
      acc += __shfl_xor(acc, 16, 32);               // combine channel halves
      if (lane < 16) out[(long)b * NVOX + v0 + (lane & 15) + 16 * s] = acc;
    }
  }
}

extern "C" void kernel_launch(void* const* d_in, const int* in_sizes, int n_in,
                              void* d_out, int out_size, void* d_ws, size_t ws_size,
                              hipStream_t stream) {
  const float* x     = (const float*)d_in[0];
  const float* d_all = (const float*)d_in[1];
  const float* W1    = (const float*)d_in[2];
  const float* W2    = (const float*)d_in[3];
  const float* G1    = (const float*)d_in[4];
  const float* G2    = (const float*)d_in[5];
  const float* G3    = (const float*)d_in[6];
  float* out = (float*)d_out;

  // 1728 blocks x 8 waves = 13824 waves; 55296 tiles -> exactly 4 tiles/wave
  dim3 grid(1728), block(256);
  hipLaunchKernelGGL(metaclf_wmma_kernel, grid, block, 0, stream,
                     x, d_all, W1, W2, G1, G2, G3, out);
}
